// MambaLiftingBlock_70033736729114
// MI455X (gfx1250) — compile-verified, matrix operand order checked by reading, not run on hardware
//
#include <hip/hip_runtime.h>
#include <hip/hip_bf16.h>
#include <cstdint>
#include <cstddef>

// ---------------------------------------------------------------------------
// Types for CDNA5 WMMA (wave32)
// ---------------------------------------------------------------------------
typedef __bf16 bf16_t;
typedef __attribute__((ext_vector_type(16))) bf16_t v16bf;
typedef __attribute__((ext_vector_type(8)))  bf16_t v8bf;
typedef __attribute__((ext_vector_type(8)))  float  v8f;
typedef __attribute__((__vector_size__(16))) int    v4i;   // async-LDS builtin arg type

#define DMODEL 1024
#define DINNER 2048
#define DSTATE 16
#define DTRANK 64
#define NXPROJ 96          // DTRANK + 2*DSTATE
#define BSZ    2
#define LSEQ   2048        // sequence length after even/odd split
#define MROWS  (BSZ*LSEQ)  // 4096 token rows per bi-mamba call

#if __has_builtin(__builtin_amdgcn_global_load_async_to_lds_b128)
#define HAVE_ASYNC_LDS 1
#else
#define HAVE_ASYNC_LDS 0
#endif

static __device__ __forceinline__ bf16_t f2bf(float f) {
  unsigned u = __float_as_uint(f);
  u += 0x7FFFu + ((u >> 16) & 1u);           // round-to-nearest-even
  unsigned short h = (unsigned short)(u >> 16);
  return __builtin_bit_cast(bf16_t, h);
}
static __device__ __forceinline__ float sigmoidf_(float x) { return 1.f / (1.f + __expf(-x)); }
static __device__ __forceinline__ float siluf_(float x)    { return x * sigmoidf_(x); }
static __device__ __forceinline__ float softplusf_(float x){ return (x > 20.f) ? x : log1pf(__expf(x)); }

// 16-byte global->LDS copy: async DMA path on gfx1250 toolchains that expose
// the builtin (emits global_load_async_to_lds_b128, tracked by ASYNCcnt),
// otherwise a register bounce (global_load_b128 + ds_store_b128).
static __device__ __forceinline__ void async_cp16(bf16_t* lds_dst, const bf16_t* g_src) {
#if HAVE_ASYNC_LDS
  __builtin_amdgcn_global_load_async_to_lds_b128((v4i*)g_src, (v4i*)lds_dst, 0, 0);
#else
  *(v8bf*)lds_dst = *(const v8bf*)g_src;
#endif
}
static __device__ __forceinline__ void wait_async_all() {
#if HAVE_ASYNC_LDS
#if __has_builtin(__builtin_amdgcn_s_wait_asynccnt)
  __builtin_amdgcn_s_wait_asynccnt(0);
#else
  asm volatile("s_wait_asynccnt 0" ::: "memory");
#endif
#endif
}

// ---------------------------------------------------------------------------
// v2 GEMM (LDS-staged, double-buffered):
//   C[M,N] = A[M,K] * W[N,K]^T (+bias) (softplus) (+residual)
// Block = 256 threads (8 waves) -> 128x64 output tile.
// Per K-step of 32: stage A(128x32) + B(64x32) bf16 panels into LDS (async),
// each wave computes a 16x64 strip = 4 WMMAs reusing one A fragment.
// Requires M%128==0, N%64==0, K%32==0.
// ---------------------------------------------------------------------------
#define LDA 40   // padded LDS row stride (bf16) to break ds bank conflicts

__global__ void k_gemm_bf16_wmma_lds(const bf16_t* __restrict__ A,
                                     const bf16_t* __restrict__ W,
                                     float* __restrict__ C,
                                     int Mr, int Nr, int Kr,
                                     const float* __restrict__ bias,
                                     const float* __restrict__ residual,
                                     int act) {
  __shared__ bf16_t sA[2][128 * LDA];
  __shared__ bf16_t sB[2][64 * LDA];

  const int tid  = threadIdx.x;
  const int lane = tid & 31;
  const int wv   = tid >> 5;            // wave id 0..7 -> M sub-tile
  const int hk   = lane >> 4;           // K-half selector (ISA 16-bit A layout)
  const int lr   = lane & 15;
  const int m0   = blockIdx.y * 128;
  const int n0   = blockIdx.x * 64;

  auto stage = [&](int buf, int k0) {
    // A panel: 128x32 bf16 = 512 16B chunks; 2 per thread
#pragma unroll
    for (int i = 0; i < 2; ++i) {
      int c = tid + 256 * i;
      int row = c >> 2, kp = (c & 3) * 8;
      async_cp16(&sA[buf][row * LDA + kp],
                 A + (size_t)(m0 + row) * Kr + k0 + kp);
    }
    // B panel: 64x32 bf16 = 256 chunks; 1 per thread
    {
      int row = tid >> 2, kp = (tid & 3) * 8;
      async_cp16(&sB[buf][row * LDA + kp],
                 W + (size_t)(n0 + row) * Kr + k0 + kp);
    }
  };

  v8f acc[4];
#pragma unroll
  for (int j = 0; j < 4; ++j)
#pragma unroll
    for (int i = 0; i < 8; ++i) acc[j][i] = 0.f;

  stage(0, 0);
  int buf = 0;
  for (int k0 = 0; k0 < Kr; k0 += 32) {
    wait_async_all();                   // this wave's staged chunks are in LDS
    __syncthreads();                    // ...and everyone else's too
    if (k0 + 32 < Kr) stage(buf ^ 1, k0 + 32);  // overlap next panel with math

    const bf16_t* pa = &sA[buf][(wv * 16 + lr) * LDA + hk * 8];
    const v8bf a0 = *(const v8bf*)pa;
    const v8bf a1 = *(const v8bf*)(pa + 16);
    v16bf av;
#pragma unroll
    for (int i = 0; i < 8; ++i) { av[i] = a0[i]; av[i + 8] = a1[i]; }

#pragma unroll
    for (int j = 0; j < 4; ++j) {
      const bf16_t* pb = &sB[buf][(j * 16 + lr) * LDA + hk * 8];
      const v8bf b0 = *(const v8bf*)pb;
      const v8bf b1 = *(const v8bf*)(pb + 16);
      v16bf bv;
#pragma unroll
      for (int i = 0; i < 8; ++i) { bv[i] = b0[i]; bv[i + 8] = b1[i]; }
      acc[j] = __builtin_amdgcn_wmma_f32_16x16x32_bf16(
          false, av, false, bv, (short)0, acc[j], false, false);
    }
    buf ^= 1;
    __syncthreads();                    // reads done before buffer is re-staged
  }

#pragma unroll
  for (int j = 0; j < 4; ++j) {
    const int col = n0 + j * 16 + lr;
#pragma unroll
    for (int r = 0; r < 8; ++r) {
      const int row = m0 + wv * 16 + hk * 8 + r;
      float v = acc[j][r];
      if (bias)     v += bias[col];
      if (act == 1) v  = softplusf_(v);
      if (residual) v += residual[(size_t)row * Nr + col];
      C[(size_t)row * Nr + col] = v;
    }
  }
}

// ---------------------------------------------------------------------------
// v1 GEMM (direct-from-global, one wave per 16x16 tile) for shapes the LDS
// kernel can't cover (x_proj: N=96).
// ---------------------------------------------------------------------------
__global__ void k_gemm_bf16_wmma(const bf16_t* __restrict__ A,
                                 const bf16_t* __restrict__ W,
                                 float* __restrict__ C,
                                 int Mr, int Nr, int Kr,
                                 const float* __restrict__ bias,
                                 const float* __restrict__ residual,
                                 int act) {
  const int lane = threadIdx.x & 31;
  const int wave = blockIdx.x * (blockDim.x >> 5) + (threadIdx.x >> 5);
  const int tn = Nr >> 4;
  const int tm = Mr >> 4;
  if (wave >= tn * tm) return;              // uniform per wave: EXEC stays all-1s
  const int tileN = wave % tn;
  const int tileM = wave / tn;

  const int hk = lane >> 4;
  const int lr = lane & 15;
  const size_t arow = (size_t)(tileM * 16 + lr) * (size_t)Kr;
  const size_t brow = (size_t)(tileN * 16 + lr) * (size_t)Kr;

  v8f acc;
#pragma unroll
  for (int i = 0; i < 8; ++i) acc[i] = 0.f;

  for (int k0 = 0; k0 < Kr; k0 += 32) {
    if (k0 + 32 < Kr) {
      __builtin_prefetch(A + arow + k0 + 32, 0, 1);
      __builtin_prefetch(W + brow + k0 + 32, 0, 1);
    }
    const v8bf a0 = *(const v8bf*)(A + arow + k0 + hk * 8);
    const v8bf a1 = *(const v8bf*)(A + arow + k0 + hk * 8 + 16);
    const v8bf b0 = *(const v8bf*)(W + brow + k0 + hk * 8);
    const v8bf b1 = *(const v8bf*)(W + brow + k0 + hk * 8 + 16);
    v16bf av, bv;
#pragma unroll
    for (int i = 0; i < 8; ++i) {
      av[i] = a0[i]; av[i + 8] = a1[i];
      bv[i] = b0[i]; bv[i + 8] = b1[i];
    }
    acc = __builtin_amdgcn_wmma_f32_16x16x32_bf16(
        false, av, false, bv, (short)0, acc, false, false);
  }

  const int col = tileN * 16 + lr;
#pragma unroll
  for (int r = 0; r < 8; ++r) {
    const int row = tileM * 16 + hk * 8 + r;
    float v = acc[r];
    if (bias)     v += bias[col];
    if (act == 1) v  = softplusf_(v);
    if (residual) v += residual[(size_t)row * Nr + col];
    C[(size_t)row * Nr + col] = v;
  }
}

// ---------------------------------------------------------------------------
// Elementwise / helper kernels
// ---------------------------------------------------------------------------
__global__ void k_split_even_odd(const float* __restrict__ x,
                                 float* __restrict__ xe, float* __restrict__ xo) {
  size_t i = (size_t)blockIdx.x * blockDim.x + threadIdx.x;
  if (i >= (size_t)MROWS * DMODEL) return;
  size_t row = i / DMODEL, col = i % DMODEL;
  size_t b = row / LSEQ, t = row % LSEQ;
  const size_t Lfull = 2 * LSEQ;
  xe[i] = x[(b * Lfull + 2 * t) * DMODEL + col];
  xo[i] = x[(b * Lfull + 2 * t + 1) * DMODEL + col];
}

__global__ void k_layernorm_bf16(const float* __restrict__ xin,
                                 const float* __restrict__ w,
                                 const float* __restrict__ b,
                                 bf16_t* __restrict__ out) {
  __shared__ float s_sum[256], s_sq[256];
  const int row = blockIdx.x;
  const float* xr = xin + (size_t)row * DMODEL;
  float s = 0.f, q = 0.f;
  for (int c = threadIdx.x; c < DMODEL; c += 256) { float v = xr[c]; s += v; q += v * v; }
  s_sum[threadIdx.x] = s; s_sq[threadIdx.x] = q;
  __syncthreads();
  for (int off = 128; off > 0; off >>= 1) {
    if ((int)threadIdx.x < off) {
      s_sum[threadIdx.x] += s_sum[threadIdx.x + off];
      s_sq[threadIdx.x]  += s_sq[threadIdx.x + off];
    }
    __syncthreads();
  }
  float mu  = s_sum[0] * (1.f / DMODEL);
  float var = s_sq[0] * (1.f / DMODEL) - mu * mu;
  float inv = rsqrtf(var + 1e-5f);
  for (int c = threadIdx.x; c < DMODEL; c += 256)
    out[(size_t)row * DMODEL + c] = f2bf((xr[c] - mu) * inv * w[c] + b[c]);
}

__global__ void k_f32_to_bf16(const float* __restrict__ in, bf16_t* __restrict__ out, size_t n) {
  size_t i = (size_t)blockIdx.x * blockDim.x + threadIdx.x;
  if (i < n) out[i] = f2bf(in[i]);
}

// Depthwise causal conv(k=4) + bias + SiLU. rev==1 reads the sequence flipped
// (everything downstream stays in the flipped time domain for the bwd branch).
__global__ void k_conv_silu(const float* __restrict__ xz,   // (M, 2*DINNER): xc | z
                            const float* __restrict__ cw,   // (DINNER,1,4)
                            const float* __restrict__ cb,   // (DINNER)
                            float* __restrict__ u,
                            bf16_t* __restrict__ u_bf,
                            int rev) {
  size_t i = (size_t)blockIdx.x * blockDim.x + threadIdx.x;
  if (i >= (size_t)MROWS * DINNER) return;
  int c = (int)(i % DINNER);
  size_t row = i / DINNER;
  int t = (int)(row % LSEQ);
  size_t b = row / LSEQ;
  float acc = cb[c];
#pragma unroll
  for (int j = 0; j < 4; ++j) {
    int s = t - 3 + j;
    if (s < 0) continue;
    int sm = rev ? (LSEQ - 1 - s) : s;
    acc += cw[c * 4 + j] * xz[((b * LSEQ + sm) * (size_t)(2 * DINNER)) + c];
  }
  float v = siluf_(acc);
  u[i] = v;
  u_bf[i] = f2bf(v);
}

__global__ void k_slice_dtr(const float* __restrict__ xdbl, bf16_t* __restrict__ dtr) {
  size_t i = (size_t)blockIdx.x * blockDim.x + threadIdx.x;
  if (i >= (size_t)MROWS * DTRANK) return;
  size_t row = i / DTRANK; int c = (int)(i % DTRANK);
  dtr[i] = f2bf(xdbl[row * NXPROJ + c]);
}

// Selective scan: one thread per (batch, channel); h[16] in registers.
// Fuses y += u*D and the silu(z) gate; emits bf16 ready for out_proj WMMA.
__global__ void k_scan(const float* __restrict__ dt,    // (M, DINNER)
                       const float* __restrict__ u,     // (M, DINNER)
                       const float* __restrict__ xdbl,  // (M, 96): B@64, C@80
                       const float* __restrict__ xz,    // (M, 2*DINNER): z at +DINNER
                       const float* __restrict__ A_log, // (DINNER, DSTATE)
                       const float* __restrict__ Dp,    // (DINNER)
                       bf16_t* __restrict__ y_bf,       // (M, DINNER)
                       int rev) {
  int idx = blockIdx.x * blockDim.x + threadIdx.x;
  if (idx >= BSZ * DINNER) return;
  int b = idx / DINNER, c = idx % DINNER;
  float a[DSTATE], h[DSTATE];
#pragma unroll
  for (int n = 0; n < DSTATE; ++n) { a[n] = -__expf(A_log[c * DSTATE + n]); h[n] = 0.f; }
  const float dcoef = Dp[c];
  for (int t = 0; t < LSEQ; ++t) {
    size_t row = (size_t)b * LSEQ + t;
    float dtv = dt[row * DINNER + c];
    float uv  = u[row * DINNER + c];
    const float* Bt = xdbl + row * NXPROJ + DTRANK;
    const float* Ct = Bt + DSTATE;
    float du = dtv * uv;
    float y = 0.f;
#pragma unroll
    for (int n = 0; n < DSTATE; ++n) {
      h[n] = h[n] * __expf(dtv * a[n]) + du * Bt[n];
      y += h[n] * Ct[n];
    }
    y += uv * dcoef;
    int torig = rev ? (LSEQ - 1 - t) : t;
    float zv = xz[((size_t)b * LSEQ + torig) * (size_t)(2 * DINNER) + DINNER + c];
    y *= siluf_(zv);
    y_bf[row * DINNER + c] = f2bf(y);
  }
}

// cat = [out_fwd | flip(out_bwd)] as bf16 for the fusion GEMM
__global__ void k_cat_flip(const float* __restrict__ of, const float* __restrict__ ob,
                           bf16_t* __restrict__ cat) {
  size_t i = (size_t)blockIdx.x * blockDim.x + threadIdx.x;
  if (i >= (size_t)MROWS * 2 * DMODEL) return;
  size_t row = i / (2 * DMODEL); int col = (int)(i % (2 * DMODEL));
  size_t b = row / LSEQ, t = row % LSEQ;
  float v;
  if (col < DMODEL) v = of[row * DMODEL + col];
  else              v = ob[((size_t)b * LSEQ + (LSEQ - 1 - t)) * DMODEL + (col - DMODEL)];
  cat[i] = f2bf(v);
}

__global__ void k_combine(const float* __restrict__ a, const float* __restrict__ b,
                          float* __restrict__ o, float sgn, size_t n) {
  size_t i = (size_t)blockIdx.x * blockDim.x + threadIdx.x;
  if (i < n) o[i] = a[i] + sgn * b[i];
}

// ---------------------------------------------------------------------------
// Host-side orchestration
// ---------------------------------------------------------------------------
struct MambaP {
  const float *A_log, *D, *conv_b, *conv_w, *dt_proj_b, *dt_proj_w,
              *in_proj_w, *out_proj_w, *x_proj_w;
};
struct BiP {
  MambaP bwd; const float *fusion_b, *fusion_w; MambaP fwd;
  const float *ln_b, *ln_w;
};
struct Ws {
  float *x_even, *x_odd, *bi_out;
  bf16_t *xn_bf;
  float *xz, *u; bf16_t *u_bf;
  float *x_dbl; bf16_t *dtr_bf;
  float *dtbuf; bf16_t *y_bf;
  float *out_fwd, *out_bwd; bf16_t *cat_bf;
  bf16_t *w_in, *w_x, *w_dt, *w_out, *w_fus;
};

static inline unsigned gblk(size_t n, unsigned b) { return (unsigned)((n + b - 1) / b); }

static void run_gemm(const bf16_t* A, const bf16_t* W, float* C,
                     int Mr, int Nr, int Kr,
                     const float* bias, const float* residual, int act,
                     hipStream_t s) {
  if ((Mr % 128 == 0) && (Nr % 64 == 0) && (Kr % 32 == 0)) {
    dim3 grid(Nr / 64, Mr / 128);
    k_gemm_bf16_wmma_lds<<<grid, 256, 0, s>>>(A, W, C, Mr, Nr, Kr, bias, residual, act);
  } else {
    int waves = (Mr / 16) * (Nr / 16);
    int blocks = (waves + 7) / 8;
    k_gemm_bf16_wmma<<<blocks, 256, 0, s>>>(A, W, C, Mr, Nr, Kr, bias, residual, act);
  }
}

static void cvt(const float* in, bf16_t* out, size_t n, hipStream_t s) {
  k_f32_to_bf16<<<gblk(n, 256), 256, 0, s>>>(in, out, n);
}

static void run_bimamba(const float* x_in, const BiP& P, float* out,
                        const Ws& w, hipStream_t s) {
  k_layernorm_bf16<<<MROWS, 256, 0, s>>>(x_in, P.ln_w, P.ln_b, w.xn_bf);
  for (int dir = 0; dir < 2; ++dir) {
    const MambaP& mp = dir ? P.bwd : P.fwd;
    cvt(mp.in_proj_w,  w.w_in,  (size_t)2 * DINNER * DMODEL, s);
    cvt(mp.x_proj_w,   w.w_x,   (size_t)NXPROJ * DINNER, s);
    cvt(mp.dt_proj_w,  w.w_dt,  (size_t)DINNER * DTRANK, s);
    cvt(mp.out_proj_w, w.w_out, (size_t)DMODEL * DINNER, s);

    run_gemm(w.xn_bf, w.w_in, w.xz, MROWS, 2 * DINNER, DMODEL, nullptr, nullptr, 0, s);
    k_conv_silu<<<gblk((size_t)MROWS * DINNER, 256), 256, 0, s>>>(
        w.xz, mp.conv_w, mp.conv_b, w.u, w.u_bf, dir);
    run_gemm(w.u_bf, w.w_x, w.x_dbl, MROWS, NXPROJ, DINNER, nullptr, nullptr, 0, s);
    k_slice_dtr<<<gblk((size_t)MROWS * DTRANK, 256), 256, 0, s>>>(w.x_dbl, w.dtr_bf);
    run_gemm(w.dtr_bf, w.w_dt, w.dtbuf, MROWS, DINNER, DTRANK,
             mp.dt_proj_b, nullptr, /*softplus*/1, s);
    k_scan<<<gblk((size_t)BSZ * DINNER, 256), 256, 0, s>>>(
        w.dtbuf, w.u, w.x_dbl, w.xz, mp.A_log, mp.D, w.y_bf, dir);
    run_gemm(w.y_bf, w.w_out, dir ? w.out_bwd : w.out_fwd,
             MROWS, DMODEL, DINNER, nullptr, nullptr, 0, s);
  }
  k_cat_flip<<<gblk((size_t)MROWS * 2 * DMODEL, 256), 256, 0, s>>>(
      w.out_fwd, w.out_bwd, w.cat_bf);
  cvt(P.fusion_w, w.w_fus, (size_t)DMODEL * 2 * DMODEL, s);
  run_gemm(w.cat_bf, w.w_fus, out, MROWS, DMODEL, 2 * DMODEL,
           P.fusion_b, /*residual=*/x_in, 0, s);
}

extern "C" void kernel_launch(void* const* d_in, const int* in_sizes, int n_in,
                              void* d_out, int out_size, void* d_ws, size_t ws_size,
                              hipStream_t stream) {
  (void)in_sizes; (void)n_in; (void)out_size; (void)ws_size;
  const float* x = (const float*)d_in[0];

  // Param leaves flattened in jax-tree (alphabetical-key) order.
  auto fillM = [&](int base) {
    MambaP m;
    m.A_log      = (const float*)d_in[base + 0];
    m.D          = (const float*)d_in[base + 1];
    m.conv_b     = (const float*)d_in[base + 2];
    m.conv_w     = (const float*)d_in[base + 3];
    m.dt_proj_b  = (const float*)d_in[base + 4];
    m.dt_proj_w  = (const float*)d_in[base + 5];
    m.in_proj_w  = (const float*)d_in[base + 6];
    m.out_proj_w = (const float*)d_in[base + 7];
    m.x_proj_w   = (const float*)d_in[base + 8];
    return m;
  };
  BiP pred, upd;
  pred.bwd = fillM(1);
  pred.fusion_b = (const float*)d_in[10];
  pred.fusion_w = (const float*)d_in[11];
  pred.fwd = fillM(12);
  pred.ln_b = (const float*)d_in[21];
  pred.ln_w = (const float*)d_in[22];
  upd.bwd = fillM(23);
  upd.fusion_b = (const float*)d_in[32];
  upd.fusion_w = (const float*)d_in[33];
  upd.fwd = fillM(34);
  upd.ln_b = (const float*)d_in[43];
  upd.ln_w = (const float*)d_in[44];

  // Workspace bump allocator (256B aligned)
  uintptr_t base = (uintptr_t)d_ws;
  auto alloc = [&](size_t bytes) -> void* {
    void* p = (void*)base;
    base += (bytes + 255) & ~(size_t)255;
    return p;
  };
  Ws w;
  w.x_even  = (float*) alloc((size_t)MROWS * DMODEL * 4);
  w.x_odd   = (float*) alloc((size_t)MROWS * DMODEL * 4);
  w.bi_out  = (float*) alloc((size_t)MROWS * DMODEL * 4);
  w.xn_bf   = (bf16_t*)alloc((size_t)MROWS * DMODEL * 2);
  w.xz      = (float*) alloc((size_t)MROWS * 2 * DINNER * 4);
  w.u       = (float*) alloc((size_t)MROWS * DINNER * 4);
  w.u_bf    = (bf16_t*)alloc((size_t)MROWS * DINNER * 2);
  w.x_dbl   = (float*) alloc((size_t)MROWS * NXPROJ * 4);
  w.dtr_bf  = (bf16_t*)alloc((size_t)MROWS * DTRANK * 2);
  w.dtbuf   = (float*) alloc((size_t)MROWS * DINNER * 4);
  w.y_bf    = (bf16_t*)alloc((size_t)MROWS * DINNER * 2);
  w.out_fwd = (float*) alloc((size_t)MROWS * DMODEL * 4);
  w.out_bwd = (float*) alloc((size_t)MROWS * DMODEL * 4);
  w.cat_bf  = (bf16_t*)alloc((size_t)MROWS * 2 * DMODEL * 2);
  w.w_in    = (bf16_t*)alloc((size_t)2 * DINNER * DMODEL * 2);
  w.w_x     = (bf16_t*)alloc((size_t)NXPROJ * DINNER * 2);
  w.w_dt    = (bf16_t*)alloc((size_t)DINNER * DTRANK * 2);
  w.w_out   = (bf16_t*)alloc((size_t)DMODEL * DINNER * 2);
  w.w_fus   = (bf16_t*)alloc((size_t)DMODEL * 2 * DMODEL * 2);

  float* s_out = (float*)d_out;                       // output 0: s
  float* d_d   = s_out + (size_t)MROWS * DMODEL;      // output 1: d (also upd input)
  const size_t nMD = (size_t)MROWS * DMODEL;

  k_split_even_odd<<<gblk(nMD, 256), 256, 0, stream>>>(x, w.x_even, w.x_odd);

  // pred = bi_mamba(x_even); d = x_odd - pred
  run_bimamba(w.x_even, pred, w.bi_out, w, stream);
  k_combine<<<gblk(nMD, 256), 256, 0, stream>>>(w.x_odd, w.bi_out, d_d, -1.f, nMD);

  // s = x_even + bi_mamba(d)
  run_bimamba(d_d, upd, w.bi_out, w, stream);
  k_combine<<<gblk(nMD, 256), 256, 0, stream>>>(w.x_even, w.bi_out, s_out, 1.f, nMD);
}
